// ClusterNetHomogeneous_33758442947292
// MI455X (gfx1250) — compile-verified, matrix operand order checked by reading, not run on hardware
//
#include <hip/hip_runtime.h>
#include <hip/hip_bf16.h>

typedef __attribute__((ext_vector_type(16))) _Float16 v16h;
typedef __attribute__((ext_vector_type(8)))  float    v8f;

#define HDIM   128
#define NGROUP 64
#define OUTD   2
#define ENC_NEG_INF 0x007FFFFFu
#define BN_RSQ 0.99999500003749969f   /* 1/sqrt(1+1e-5) */

// Monotonic bijection float -> u32 so that unsigned-int max == float max.
static __device__ __forceinline__ unsigned enc_f32(float f) {
  unsigned u = __float_as_uint(f);
  return (u & 0x80000000u) ? ~u : (u | 0x80000000u);
}
static __device__ __forceinline__ float dec_f32(unsigned u) {
  unsigned b = (u & 0x80000000u) ? (u ^ 0x80000000u) : ~u;
  return __uint_as_float(b);
}

// ---------------- layer input build (concat + agg init), dense stride C
__global__ void build_h0(const float* __restrict__ x, const float* __restrict__ pos,
                         float* __restrict__ h, unsigned* __restrict__ agg,
                         int N, int C) {
  int i = blockIdx.x;
  if (i >= N) return;
  for (int c = threadIdx.x; c < C; c += blockDim.x) {
    float v = (c < 64) ? x[(size_t)i * 64 + c] : pos[(size_t)i * 2 + (c - 64)];
    h[(size_t)i * C + c]   = v;
    agg[(size_t)i * C + c] = ENC_NEG_INF;
  }
}

__global__ void build_h_mid(const float* __restrict__ hin, const float* __restrict__ pos,
                            float* __restrict__ h, unsigned* __restrict__ agg,
                            int N, int C) {
  int i = blockIdx.x;
  if (i >= N) return;
  for (int c = threadIdx.x; c < C; c += blockDim.x) {
    float v = (c < 128) ? hin[(size_t)i * 128 + c] : pos[(size_t)i * 2 + (c - 128)];
    h[(size_t)i * C + c]   = v;
    agg[(size_t)i * C + c] = ENC_NEG_INF;
  }
}

// ---------------- edge scatter-max (feature-parallel, coalesced) -------
// Dense stride C keeps the (h, agg) working set inside the 192MB L2.
__global__ void scatter_max_kernel(const int* __restrict__ src, const int* __restrict__ dst,
                                   const float* __restrict__ h, unsigned* __restrict__ agg,
                                   int E, int C) {
  int e = blockIdx.x * 2 + (threadIdx.x >> 7);   // 2 edges per 256-thread block
  if (e >= E) return;
  int c0 = threadIdx.x & 127;
  int s = src[e], d = dst[e];
  const float* hs = h + (size_t)s * C;
  unsigned*    ad = agg + (size_t)d * C;
  for (int c = c0; c < C; c += 128) {
    atomicMax(&ad[c], enc_f32(hs[c]));
  }
}

// ---------------- comb = h + agg (neg-inf -> 0), cast f16, zero-pad ----
__global__ void prep_a(const float* __restrict__ h, const unsigned* __restrict__ agg,
                       _Float16* __restrict__ A, int N, int C, int Kpad) {
  int i = blockIdx.x;           // row
  int c = threadIdx.x;          // blockDim == Kpad (96 or 160)
  if (i >= N || c >= Kpad) return;
  float v = 0.f;
  if (c < C) {
    float a = dec_f32(agg[(size_t)i * C + c]);
    if (a == -__builtin_inff()) a = 0.f;   // empty neighborhood (PyG fill)
    v = h[(size_t)i * C + c] + a;
  }
  A[(size_t)i * Kpad + c] = (_Float16)v;
}

// ---------------- weight pack into WMMA B-fragment order ---------------
// frag index idx = (kt*8 + j)*32 + lane, 16 halves each, matching the
// 16-bit 32x16 B VGPR layout (ISA 7.12.2).
__global__ void pack_b_kernel(const float* __restrict__ W, _Float16* __restrict__ Bp,
                              int Kin, int Kpad) {
  int idx = blockIdx.x * blockDim.x + threadIdx.x;
  int total = (Kpad >> 5) * 8 * 32;
  if (idx >= total) return;
  int lane = idx & 31;
  int j    = (idx >> 5) & 7;
  int kt   = idx >> 8;
  int n     = j * 16 + (lane & 15);
  int kbase = kt * 32 + ((lane & 16) ? 8 : 0);
  _Float16* out = Bp + (size_t)idx * 16;
#pragma unroll
  for (int v = 0; v < 8; ++v) {
    int k = kbase + ((v < 4) ? 0 : 16) + 2 * (v & 3);
    out[2 * v]     = (k     < Kin) ? (_Float16)W[(size_t)k * HDIM + n]       : (_Float16)0.f;
    out[2 * v + 1] = (k + 1 < Kin) ? (_Float16)W[(size_t)(k + 1) * HDIM + n] : (_Float16)0.f;
  }
}

// ---------------- fused GEMM + bias + BN + ReLU (WMMA f16->f32) --------
// Each wave: 16 rows x 128 cols.  Block = 8 waves = 128 rows.
// Packed weights are staged once per block into LDS (<=40KB of the WGP's
// 320KB), so B-fragment reads are short-latency ds_load_b128 shared by
// all 8 waves instead of per-WMMA global loads.
__global__ __launch_bounds__(256, 1) void gin_gemm(
    const _Float16* __restrict__ A, const _Float16* __restrict__ Bp,
    const float* __restrict__ bias, const float* __restrict__ gamma,
    const float* __restrict__ beta,
    _Float16* __restrict__ out16, float* __restrict__ out32,
    int M, int Kpad) {
  extern __shared__ _Float16 Bs[];

  // Cooperative copy of the whole packed B (Kpad*128 halves = Kpad*16 uint4)
  {
    const int nchunk = Kpad * 16;
    const uint4* __restrict__ src = (const uint4*)Bp;
    uint4* dstp = (uint4*)Bs;
    for (int idx = threadIdx.x; idx < nchunk; idx += 256)
      dstp[idx] = src[idx];
  }
  __syncthreads();

  const int lane = threadIdx.x & 31;
  const int wave = threadIdx.x >> 5;
  const int tile = blockIdx.x * 8 + wave;
  const int m0 = tile * 16;
  if (m0 < M) {
    const int mrow = m0 + (lane & 15);
    const int arow = (mrow < M) ? mrow : (M - 1);
    const _Float16* ap = A + (size_t)arow * Kpad;
    const int koff = (lane & 16) ? 8 : 0;

    v8f zero = {0.f, 0.f, 0.f, 0.f, 0.f, 0.f, 0.f, 0.f};
    v8f acc[8];
#pragma unroll
    for (int j = 0; j < 8; ++j) acc[j] = zero;

    for (int kk = 0; kk < Kpad; kk += 32) {
      // A fragment: 16-bit 16x32 layout — per-lane K pairs (ISA 7.12.2);
      // pairs are contiguous 16B chunks, so these merge into 2x b128 loads.
      v16h a;
#pragma unroll
      for (int v = 0; v < 8; ++v) {
        int k = kk + koff + ((v < 4) ? 0 : 16) + 2 * (v & 3);
        a[2 * v]     = ap[k];
        a[2 * v + 1] = ap[k + 1];
      }

      const _Float16* bp = Bs + ((size_t)(kk >> 5) * 8 * 32 + lane) * 16;
      v16h b[8];
#pragma unroll
      for (int j = 0; j < 8; ++j)
        b[j] = *(const v16h*)(bp + (size_t)j * 32 * 16);

#pragma unroll
      for (int j = 0; j < 8; ++j) {
        acc[j] = __builtin_amdgcn_wmma_f32_16x16x32_f16(
            /*neg_a=*/false, a, /*neg_b=*/false, b[j],
            /*c_mod=*/(short)0, acc[j], /*reuse_a=*/false, /*reuse_b=*/false);
      }
    }

    // Epilogue: C/D layout — VGPR r: lanes0-15 -> M=r, lanes16-31 -> M=r+8
    const int nlo = lane & 15;
    const int mhi = (lane >> 4) & 1;
#pragma unroll
    for (int j = 0; j < 8; ++j) {
      int col = j * 16 + nlo;
      float bcol = bias[col];
      float scol = gamma[col] * BN_RSQ;
      float shft = beta[col];
#pragma unroll
      for (int r = 0; r < 8; ++r) {
        int row = m0 + r + mhi * 8;
        if (row >= M) continue;
        float v = (acc[j][r] + bcol) * scol + shft;
        v = v > 0.f ? v : 0.f;
        if (out32) out32[(size_t)row * HDIM + col] = v;
        if (out16) out16[(size_t)row * HDIM + col] = (_Float16)v;
      }
    }
  }
}

// ---------------- global max pool + final linear -----------------------
__global__ void pool_init(unsigned* __restrict__ poolb) {
  int i = blockIdx.x * blockDim.x + threadIdx.x;
  if (i < NGROUP * HDIM) poolb[i] = ENC_NEG_INF;
}

__global__ void pool_scatter(const float* __restrict__ h, const int* __restrict__ batch,
                             unsigned* __restrict__ poolb, int N) {
  int i = blockIdx.x;
  if (i >= N) return;
  int c = threadIdx.x;  // 128 threads
  atomicMax(&poolb[(size_t)batch[i] * HDIM + c], enc_f32(h[(size_t)i * HDIM + c]));
}

__global__ void final_linear(const unsigned* __restrict__ poolb,
                             const float* __restrict__ wlin, const float* __restrict__ blin,
                             float* __restrict__ out) {
  int t = threadIdx.x;
  if (t >= NGROUP * OUTD) return;
  int g = t / OUTD, o = t % OUTD;
  float s = blin[o];
  for (int c = 0; c < HDIM; ++c) {
    float v = dec_f32(poolb[g * HDIM + c]);
    if (v == -__builtin_inff()) v = 0.f;
    s += v * wlin[c * OUTD + o];
  }
  out[t] = s;
}

// ---------------- host launch -----------------------------------------
extern "C" void kernel_launch(void* const* d_in, const int* in_sizes, int n_in,
                              void* d_out, int out_size, void* d_ws, size_t ws_size,
                              hipStream_t stream) {
  (void)n_in; (void)out_size; (void)ws_size;
  const float* x     = (const float*)d_in[0];
  const float* pos   = (const float*)d_in[1];
  const int*   ei    = (const int*)d_in[2];
  const int*   batch = (const int*)d_in[3];
  const float* wlin  = (const float*)d_in[28];
  const float* blin  = (const float*)d_in[29];

  const int N = in_sizes[0] / 64;
  const int E = in_sizes[2] / 2;
  const int* src = ei;
  const int* dst = ei + E;

  // Workspace carve-out
  const int CMAX    = 130;
  const int KPADMAX = 160;
  char* wsp = (char*)d_ws;
  size_t off = 0;
  auto carve = [&](size_t bytes) -> char* {
    char* p = wsp + off;
    off += (bytes + 255) & ~(size_t)255;
    return p;
  };
  float*    hbuf  = (float*)    carve((size_t)N * CMAX * 4);
  unsigned* agg   = (unsigned*) carve((size_t)N * CMAX * 4);
  _Float16* A16   = (_Float16*) carve((size_t)N * KPADMAX * 2);
  _Float16* mid16 = (_Float16*) carve((size_t)N * HDIM * 2);
  float*    hout  = (float*)    carve((size_t)N * HDIM * 4);
  _Float16* w1p   = (_Float16*) carve((size_t)KPADMAX * HDIM * 2);
  _Float16* w2p   = (_Float16*) carve((size_t)HDIM * HDIM * 2);
  unsigned* poolb = (unsigned*) carve((size_t)NGROUP * HDIM * 4);

  const int mtiles = (N + 15) / 16;
  const int gemm_blocks = (mtiles + 7) / 8;

  for (int l = 0; l < 3; ++l) {
    const int C    = (l == 0) ? 66 : 130;
    const int Kpad = (l == 0) ? 96 : 160;
    const float* w1  = (const float*)d_in[4 + l * 8 + 0];
    const float* b1  = (const float*)d_in[4 + l * 8 + 1];
    const float* g1  = (const float*)d_in[4 + l * 8 + 2];
    const float* be1 = (const float*)d_in[4 + l * 8 + 3];
    const float* w2  = (const float*)d_in[4 + l * 8 + 4];
    const float* b2  = (const float*)d_in[4 + l * 8 + 5];
    const float* g2  = (const float*)d_in[4 + l * 8 + 6];
    const float* be2 = (const float*)d_in[4 + l * 8 + 7];

    const int pb1 = (Kpad / 32) * 8 * 32;
    pack_b_kernel<<<(pb1 + 255) / 256, 256, 0, stream>>>(w1, w1p, C, Kpad);
    const int pb2 = (HDIM / 32) * 8 * 32;
    pack_b_kernel<<<(pb2 + 255) / 256, 256, 0, stream>>>(w2, w2p, HDIM, HDIM);

    if (l == 0)
      build_h0<<<N, 128, 0, stream>>>(x, pos, hbuf, agg, N, C);
    else
      build_h_mid<<<N, 160, 0, stream>>>(hout, pos, hbuf, agg, N, C);

    scatter_max_kernel<<<(E + 1) / 2, 256, 0, stream>>>(src, dst, hbuf, agg, E, C);

    prep_a<<<N, Kpad, 0, stream>>>(hbuf, agg, A16, N, C, Kpad);

    gin_gemm<<<gemm_blocks, 256, (size_t)Kpad * 256, stream>>>(
        A16, w1p, b1, g1, be1, mid16, (float*)nullptr, N, Kpad);
    gin_gemm<<<gemm_blocks, 256, (size_t)HDIM * 256, stream>>>(
        mid16, w2p, b2, g2, be2, (_Float16*)nullptr, hout, N, HDIM);
  }

  pool_init<<<(NGROUP * HDIM + 255) / 256, 256, 0, stream>>>(poolb);
  pool_scatter<<<N, HDIM, 0, stream>>>(hout, batch, poolb, N);
  final_linear<<<1, 128, 0, stream>>>(poolb, wlin, blin, (float*)d_out);
}
